// AttentionBlock_9706626089732
// MI455X (gfx1250) — matrix operand order repaired and checked by reading, not compiled, not run on hardware
//
#include <hip/hip_runtime.h>
#include <stdint.h>

// ---------------------------------------------------------------------------
// CDNA5 (gfx1250) wave32 WMMA types & helpers
// ---------------------------------------------------------------------------
typedef __attribute__((ext_vector_type(16))) __bf16 v16bf;
typedef __attribute__((ext_vector_type(8)))  float  v8f;
typedef __attribute__((ext_vector_type(8)))  unsigned short us8;

union Frag {
  v16bf bf;
  us8   u8[2];
  unsigned short us[16];
};

__device__ __forceinline__ unsigned short f32_to_bf16(float f) {
  union { float f; unsigned u; } v; v.f = f;
  unsigned u = v.u;
  u += 0x7FFFu + ((u >> 16) & 1u);          // round-to-nearest-even
  return (unsigned short)(u >> 16);
}

__device__ __forceinline__ v8f wmma_bf16(const Frag& a, const Frag& b, v8f c) {
  // D = A(16x32 bf16) * B(32x16 bf16) + C(16x16 f32)
  return __builtin_amdgcn_wmma_f32_16x16x32_bf16(
      /*neg_a=*/false, a.bf, /*neg_b=*/false, b.bf,
      /*c_mod=*/(short)0, c, /*reuse_a=*/false, /*reuse_b=*/false);
}

// ---------------------------------------------------------------------------
// 0) f32 -> bf16 cast (weights / context)
// ---------------------------------------------------------------------------
__global__ void cast_bf16_kernel(const float* __restrict__ src,
                                 unsigned short* __restrict__ dst, int n) {
  int idx = blockIdx.x * 256 + threadIdx.x;
  if (idx < n) dst[idx] = f32_to_bf16(src[idx]);
}

// ---------------------------------------------------------------------------
// 1) GroupNorm over (16 ch x 1024 hw) per (b, group); writes xn TRANSPOSED
//    as bf16 [b][hw=1024][c=512] so the qkv GEMM reads K-contiguous rows.
// ---------------------------------------------------------------------------
__global__ void groupnorm_kernel(const float* __restrict__ x,
                                 const float* __restrict__ gw,
                                 const float* __restrict__ gb,
                                 unsigned short* __restrict__ xnT) {
  __shared__ float s1[256], s2[256];
  int b = blockIdx.x >> 5;   // 32 groups
  int g = blockIdx.x & 31;
  const float* xp = x + ((size_t)b * 512 + g * 16) * 1024;
  float sum = 0.f, sq = 0.f;
  for (int e = threadIdx.x; e < 16384; e += 256) {
    float v = xp[e]; sum += v; sq += v * v;
  }
  s1[threadIdx.x] = sum; s2[threadIdx.x] = sq;
  __syncthreads();
  for (int s = 128; s > 0; s >>= 1) {
    if (threadIdx.x < s) {
      s1[threadIdx.x] += s1[threadIdx.x + s];
      s2[threadIdx.x] += s2[threadIdx.x + s];
    }
    __syncthreads();
  }
  float mean = s1[0] * (1.0f / 16384.0f);
  float var  = s2[0] * (1.0f / 16384.0f) - mean * mean;
  float rstd = rsqrtf(var + 1e-5f);
  for (int e = threadIdx.x; e < 16384; e += 256) {
    int cl = e >> 10, i = e & 1023;
    int c = g * 16 + cl;
    float vn = (xp[e] - mean) * rstd * gw[c] + gb[c];
    xnT[((size_t)b * 1024 + i) * 512 + c] = f32_to_bf16(vn);
  }
}

// ---------------------------------------------------------------------------
// 2) Batched bf16 WMMA GEMM: C[b][M][N] = A[b][M][K] * W[N][K]^T + bias[N]
//    Register-blocked: each wave computes a 16M x 64N strip (4 accumulators),
//    so the A fragment is loaded once per k-step and reused for 4 WMMAs
//    (40 B/lane per WMMA instead of 64). W is the natural (out,in) weight
//    layout so B-fragments load as two contiguous b128s per lane.
//    grid: (N/256, ceil(M/16), batch), block 128 (4 waves, 64 N-cols each).
// ---------------------------------------------------------------------------
__global__ void gemm_bf16_kernel(const unsigned short* __restrict__ A, long aStride,
                                 int M, int K,
                                 const unsigned short* __restrict__ W,
                                 const float* __restrict__ bias,
                                 float* __restrict__ C, long cStride, int N) {
  int lane = threadIdx.x & 31, wave = threadIdx.x >> 5;
  int lm = lane & 15, hi = lane >> 4;
  int n0 = (blockIdx.x * 4 + wave) * 64;
  int m0 = blockIdx.y * 16;
  A += (size_t)blockIdx.z * aStride;
  C += (size_t)blockIdx.z * cStride;
  int row = m0 + lm; if (row > M - 1) row = M - 1;   // clamp edge reads
  const unsigned short* arow = A + (size_t)row * K + hi * 8;          // A kOff
  const unsigned short* w0   = W + (size_t)(n0 + lm) * K + hi * 8;    // B kOff
  const long wstride = (long)16 * K;
  v8f z = {0, 0, 0, 0, 0, 0, 0, 0};
  v8f acc[4] = {z, z, z, z};
  for (int k = 0; k < K; k += 32) {
    Frag a;
    a.u8[0] = *(const us8*)(arow + k);
    a.u8[1] = *(const us8*)(arow + k + 16);
#pragma unroll
    for (int t = 0; t < 4; ++t) {
      const unsigned short* wrow = w0 + (size_t)t * wstride + k;
      Frag bf;
      bf.u8[0] = *(const us8*)(wrow);
      bf.u8[1] = *(const us8*)(wrow + 16);
      acc[t] = wmma_bf16(a, bf, acc[t]);
    }
  }
#pragma unroll
  for (int t = 0; t < 4; ++t) {
    float bv = bias ? bias[n0 + t * 16 + lm] : 0.0f;
#pragma unroll
    for (int r = 0; r < 8; ++r) {
      int mr = m0 + r + 8 * hi;
      if (mr < M) C[(size_t)mr * N + n0 + t * 16 + lm] = acc[t][r] + bv;
    }
  }
}

// ---------------------------------------------------------------------------
// 3) Split qkv GEMM output into Q (pre-scaled x0.125), K rows [0,1024),
//    V cols [0,1024) in attention layouts (bf16).
//    Q: [bn][1024][64], K: [bn][1120][64], V: [bn][64][1120]
// ---------------------------------------------------------------------------
__global__ void assemble_self_kernel(const float* __restrict__ Cqkv,
                                     unsigned short* __restrict__ Qb,
                                     unsigned short* __restrict__ Kb,
                                     unsigned short* __restrict__ Vb) {
  int idx = blockIdx.x * 256 + threadIdx.x;    // 64*1024*64 total
  int d  = idx & 63;
  int i  = (idx >> 6) & 1023;
  int bn = idx >> 16;                          // b*8 + n
  int n = bn & 7, b = bn >> 3;
  const float* crow = Cqkv + ((size_t)b * 1024 + i) * 1536 + n * 64 + d;
  float q = crow[0], k = crow[512], v = crow[1024];
  Qb[((size_t)bn * 1024 + i) * 64 + d] = f32_to_bf16(q * 0.125f); // scale^2
  Kb[((size_t)bn * 1120 + i) * 64 + d] = f32_to_bf16(k);
  Vb[((size_t)bn * 64 + d) * 1120 + i] = f32_to_bf16(v);
}

// Context part: K/V columns j in [1024, 1120); zero-pad beyond L=77.
__global__ void assemble_ctx_kernel(const float* __restrict__ Cckv,
                                    unsigned short* __restrict__ Kb,
                                    unsigned short* __restrict__ Vb) {
  int idx = blockIdx.x * 256 + threadIdx.x;    // 64*96*64 total
  int d  = idx & 63;
  int t  = idx >> 6;
  int jj = t % 96;
  int bn = t / 96;
  int n = bn & 7, b = bn >> 3;
  float k = 0.f, v = 0.f;
  if (jj < 77) {
    const float* crow = Cckv + ((size_t)b * 77 + jj) * 2048 + n * 64 + d;
    k = crow[0]; v = crow[512];
  }
  Kb[((size_t)bn * 1120 + 1024 + jj) * 64 + d] = f32_to_bf16(k);
  Vb[((size_t)bn * 64 + d) * 1120 + 1024 + jj] = f32_to_bf16(v);
}

// ---------------------------------------------------------------------------
// 4) Fused attention. Block = 128 thr (4 waves); each wave owns 16 query
//    rows. Full 16x1120 f32 score panel per wave lives in LDS (287KB/block,
//    fits the 320KB WGP LDS) -> exact fp32 softmax, no online rescaling.
//    QK^T and P*V are bf16 WMMA. Output written transposed: oT[b][i][c].
// ---------------------------------------------------------------------------
__global__ void flash_attn_kernel(const unsigned short* __restrict__ Qb,
                                  const unsigned short* __restrict__ Kb,
                                  const unsigned short* __restrict__ Vb,
                                  const int* __restrict__ mask,
                                  unsigned short* __restrict__ oT) {
  extern __shared__ float smem[];              // 4*16*1120 floats + 64 floats
  int lane = threadIdx.x & 31, wave = threadIdx.x >> 5;
  int lm = lane & 15, hi = lane >> 4;
  int bn = blockIdx.y, b = bn >> 3, n = bn & 7;
  int i0 = blockIdx.x * 64 + wave * 16;
  float* S    = smem + (size_t)wave * 16 * 1120;
  float* invb = smem + 4 * 16 * 1120;

  const unsigned short* Q  = Qb + (size_t)bn * 1024 * 64;
  const unsigned short* Kp = Kb + (size_t)bn * 1120 * 64;
  const unsigned short* Vp = Vb + (size_t)bn * 64 * 1120;

  // Preload Q fragments for K=64 (two 32-deep steps); reused across 70 tiles
  const unsigned short* qrow = Q + (size_t)(i0 + lm) * 64 + hi * 8;
  Frag aq0, aq1;
  aq0.u8[0] = *(const us8*)(qrow + 0);
  aq0.u8[1] = *(const us8*)(qrow + 16);
  aq1.u8[0] = *(const us8*)(qrow + 32);
  aq1.u8[1] = *(const us8*)(qrow + 48);

  // S = Q K^T over 70 j-tiles of 16
  for (int jt = 0; jt < 70; ++jt) {
    int j0 = jt * 16;
    const unsigned short* krow = Kp + (size_t)(j0 + lm) * 64 + hi * 8;
    Frag bk; v8f acc = {0, 0, 0, 0, 0, 0, 0, 0};
    bk.u8[0] = *(const us8*)(krow + 0);
    bk.u8[1] = *(const us8*)(krow + 16);
    acc = wmma_bf16(aq0, bk, acc);
    bk.u8[0] = *(const us8*)(krow + 32);
    bk.u8[1] = *(const us8*)(krow + 48);
    acc = wmma_bf16(aq1, bk, acc);
#pragma unroll
    for (int r = 0; r < 8; ++r)
      S[(r + 8 * hi) * 1120 + j0 + lm] = acc[r];
  }
  __syncthreads();

  // Mask context columns (j >= 1024): invalid where padded or mask==0
  for (int jj = lane; jj < 96; jj += 32) {
    bool ok = (jj < 77) && (mask[b * 77 + jj] != 0);
    if (!ok) {
#pragma unroll
      for (int r = 0; r < 16; ++r) S[r * 1120 + 1024 + jj] = -1e30f;
    }
  }
  __syncthreads();

  // Exact fp32 softmax: lane handles row lm, half (hi) of 1120 columns;
  // combine halves via wave32 shuffle across lane^16.
  {
    float* Srow = S + lm * 1120;
    int j0 = hi * 560;
    float mx = -1e30f;
    for (int j = 0; j < 560; ++j) mx = fmaxf(mx, Srow[j0 + j]);
    mx = fmaxf(mx, __shfl_xor(mx, 16, 32));
    float l = 0.0f;
    for (int j = 0; j < 560; ++j) {
      float e = __expf(Srow[j0 + j] - mx);
      Srow[j0 + j] = e;
      l += e;
    }
    l += __shfl_xor(l, 16, 32);
    if (hi == 0) invb[wave * 16 + lm] = 1.0f / l;
  }
  __syncthreads();

  // O[16][64] = P[16][1120] * V^T ; 4 d-tiles, 35 K-steps of 32
  v8f z = {0, 0, 0, 0, 0, 0, 0, 0};
  v8f oacc[4] = {z, z, z, z};
  for (int ks = 0; ks < 35; ++ks) {
    int kb = ks * 32;
    const float* prow = S + lm * 1120 + kb + hi * 8;
    Frag ap;
#pragma unroll
    for (int e = 0; e < 8; ++e) ap.us[e]     = f32_to_bf16(prow[e]);
#pragma unroll
    for (int e = 0; e < 8; ++e) ap.us[8 + e] = f32_to_bf16(prow[16 + e]);
#pragma unroll
    for (int dt = 0; dt < 4; ++dt) {
      const unsigned short* vrow = Vp + (size_t)(dt * 16 + lm) * 1120 + kb + hi * 8;
      Frag bv;
      bv.u8[0] = *(const us8*)(vrow + 0);
      bv.u8[1] = *(const us8*)(vrow + 16);
      oacc[dt] = wmma_bf16(ap, bv, oacc[dt]);
    }
  }

  // Normalize rows and store bf16, transposed for the proj GEMM
#pragma unroll
  for (int dt = 0; dt < 4; ++dt) {
#pragma unroll
    for (int r = 0; r < 8; ++r) {
      int irow = i0 + r + 8 * hi;
      float val = oacc[dt][r] * invb[wave * 16 + r + 8 * hi];
      oT[((size_t)b * 1024 + irow) * 512 + n * 64 + dt * 16 + lm] = f32_to_bf16(val);
    }
  }
}

// ---------------------------------------------------------------------------
// 5) out[b][c][i] = Cproj[b][i][c] + x[b][c][i]
// ---------------------------------------------------------------------------
__global__ void residual_kernel(const float* __restrict__ Cproj,
                                const float* __restrict__ x,
                                float* __restrict__ out) {
  int idx = blockIdx.x * 256 + threadIdx.x;    // 8*512*1024 total
  int i = idx & 1023;
  int o = (idx >> 10) & 511;
  int b = idx >> 19;
  out[idx] = Cproj[((size_t)b * 1024 + i) * 512 + o] + x[idx];
}

// ---------------------------------------------------------------------------
extern "C" void kernel_launch(void* const* d_in, const int* in_sizes, int n_in,
                              void* d_out, int out_size, void* d_ws, size_t ws_size,
                              hipStream_t stream) {
  const float* x       = (const float*)d_in[0];
  const float* context = (const float*)d_in[1];
  const float* gn_w    = (const float*)d_in[2];
  const float* gn_b    = (const float*)d_in[3];
  const float* qkv_w   = (const float*)d_in[4];
  const float* qkv_b   = (const float*)d_in[5];
  const float* ckv_w   = (const float*)d_in[6];
  const float* ckv_b   = (const float*)d_in[7];
  const float* proj_w  = (const float*)d_in[8];
  const float* proj_b  = (const float*)d_in[9];
  const int*   mask    = (const int*)d_in[10];
  float* out = (float*)d_out;

  // workspace carve-out (~107 MB total), 256B aligned
  char* ws = (char*)d_ws;
  size_t off = 0;
  auto alloc = [&](size_t bytes) -> char* {
    char* p = ws + off;
    off += (bytes + 255) & ~(size_t)255;
    return p;
  };
  unsigned short* wq    = (unsigned short*)alloc((size_t)1536 * 512 * 2);
  unsigned short* wckv  = (unsigned short*)alloc((size_t)2048 * 1024 * 2);
  unsigned short* wproj = (unsigned short*)alloc((size_t)512 * 512 * 2);
  unsigned short* ctxbf = (unsigned short*)alloc((size_t)8 * 77 * 1024 * 2);
  unsigned short* xnT   = (unsigned short*)alloc((size_t)8 * 1024 * 512 * 2);
  float* Cqkv           = (float*)alloc((size_t)8 * 1024 * 1536 * 4);
  float* Cckv           = (float*)alloc((size_t)8 * 77 * 2048 * 4);
  unsigned short* Qb    = (unsigned short*)alloc((size_t)64 * 1024 * 64 * 2);
  unsigned short* Kb    = (unsigned short*)alloc((size_t)64 * 1120 * 64 * 2);
  unsigned short* Vb    = (unsigned short*)alloc((size_t)64 * 64 * 1120 * 2);
  unsigned short* oTb   = (unsigned short*)alloc((size_t)8 * 1024 * 512 * 2);
  float* Cproj = Cqkv;  // Cqkv is dead after assemble_self -> reuse for proj

  cast_bf16_kernel<<<(786432 + 255) / 256, 256, 0, stream>>>(qkv_w, wq, 786432);
  cast_bf16_kernel<<<(2097152 + 255) / 256, 256, 0, stream>>>(ckv_w, wckv, 2097152);
  cast_bf16_kernel<<<(262144 + 255) / 256, 256, 0, stream>>>(proj_w, wproj, 262144);
  cast_bf16_kernel<<<(630784 + 255) / 256, 256, 0, stream>>>(context, ctxbf, 630784);

  groupnorm_kernel<<<256, 256, 0, stream>>>(x, gn_w, gn_b, xnT);

  // qkv: per-batch [1024 x 1536] = xnT[1024 x 512] * qkv_w^T
  gemm_bf16_kernel<<<dim3(6, 64, 8), 128, 0, stream>>>(
      xnT, (long)1024 * 512, 1024, 512, wq, qkv_b, Cqkv, (long)1024 * 1536, 1536);
  // ckv: per-batch [77 x 2048] = context[77 x 1024] * ckv_w^T
  gemm_bf16_kernel<<<dim3(8, 5, 8), 128, 0, stream>>>(
      ctxbf, (long)77 * 1024, 77, 1024, wckv, ckv_b, Cckv, (long)77 * 2048, 2048);

  assemble_self_kernel<<<16384, 256, 0, stream>>>(Cqkv, Qb, Kb, Vb);
  assemble_ctx_kernel<<<1536, 256, 0, stream>>>(Cckv, Kb, Vb);

  size_t lds_bytes = (size_t)4 * 16 * 1120 * 4 + 64 * 4;   // 287 KB < 320 KB
  flash_attn_kernel<<<dim3(16, 64), 128, lds_bytes, stream>>>(Qb, Kb, Vb, mask, oTb);

  // proj: per-batch [1024 x 512] = oT[1024 x 512] * proj_w^T
  gemm_bf16_kernel<<<dim3(2, 64, 8), 128, 0, stream>>>(
      oTb, (long)1024 * 512, 1024, 512, wproj, proj_b, Cproj, (long)1024 * 512, 512);

  residual_kernel<<<16384, 256, 0, stream>>>(Cproj, x, out);
}
